// GAT_77180562309135
// MI455X (gfx1250) — matrix operand order repaired and checked by reading, not compiled, not run on hardware
//
#include <hip/hip_runtime.h>
#include <hip/hip_bf16.h>
#include <math.h>
#include <stdint.h>

// ---------------------------------------------------------------------------
// GATv2 GNN forward for MI455X (gfx1250), wave32.
// GEMMs: f16 WMMA (f32 accumulate), register-blocked 16xKout per wave,
// W^T staged into LDS by the Tensor Data Mover (tensor_load_to_lds).
// ---------------------------------------------------------------------------

typedef __attribute__((ext_vector_type(16))) _Float16 v16h;
typedef __attribute__((ext_vector_type(8)))  float    v8f;
typedef __attribute__((ext_vector_type(4)))  unsigned int v4u;
typedef __attribute__((ext_vector_type(8)))  unsigned int v8u;

#define NNODES 50000
#define NEDGES 400000
#define NGRAPH 512
#define ROWTILES (NNODES / 16)   // 3125
#define LRELU(z) ((z) > 0.0f ? (z) : 0.2f * (z))

// -------------------------------- fill -------------------------------------
__global__ void gat_fill_kernel(float* __restrict__ p, float val, long n) {
    long i = (long)blockIdx.x * blockDim.x + threadIdx.x;
    if (i < n) p[i] = val;
}

// -------------------------- f32 -> f16 convert ------------------------------
__global__ void gat_f16cvt_kernel(const float* __restrict__ in,
                                  _Float16* __restrict__ out, long n) {
    long i = (long)blockIdx.x * blockDim.x + threadIdx.x;
    if (i < n) out[i] = (_Float16)in[i];
}

// W[Kin x Kout] (f32) -> Wt[Kout x Kin] (f16)
__global__ void gat_wt_f16_kernel(const float* __restrict__ W,
                                  _Float16* __restrict__ Wt, int Kin, int Kout) {
    long i = (long)blockIdx.x * blockDim.x + threadIdx.x;
    if (i >= (long)Kin * Kout) return;
    int r = (int)(i / Kout), c = (int)(i % Kout);
    Wt[(size_t)c * Kin + r] = (_Float16)W[i];
}

// ---------------------------- WMMA GEMM ------------------------------------
// C[N x Kout] = Xh[N x Kin] @ Wt^T, Xh f16 row-major, Wt f16 [Kout x Kin].
// Block = 256 threads = 8 waves; each wave computes 16 rows x Kout cols.
// Wt staged into LDS once per block by the TDM (tensor_load_to_lds).
template <int NCT>   // NCT = Kout/16
__global__ void __launch_bounds__(256)
gat_gemm_wmma_kernel(const _Float16* __restrict__ Xh,
                     const _Float16* __restrict__ Wt,
                     float* __restrict__ C, int Kin) {
    constexpr int Kout = NCT * 16;
    __shared__ _Float16 sW[32768];   // up to 256x128 f16 = 64 KB

    // --- TDM: DMA Wt (Kout x Kin f16, 2D tensor) into LDS, wave 0 only ---
    if ((threadIdx.x >> 5) == 0) {
        unsigned long long ga = (unsigned long long)(uintptr_t)Wt;
        unsigned lofs = (unsigned)(uintptr_t)(&sW[0]);   // flat low 32 = LDS offset
        unsigned kin = (unsigned)Kin, kout = (unsigned)Kout;
        v4u g0;
        g0[0] = 1u;                                    // count=1, user mode
        g0[1] = lofs;                                  // lds_addr
        g0[2] = (unsigned)ga;                          // global_addr[31:0]
        g0[3] = ((unsigned)(ga >> 32) & 0x01FFFFFFu)   // global_addr[56:32]
              | (2u << 30);                            // type = 2 (image)
        v8u g1;
        g1[0] = 1u << 16;                              // data_size = 2 bytes
        g1[1] = (kin & 0xFFFFu) << 16;                 // tensor_dim0[15:0]
        g1[2] = (kin >> 16) | ((kout & 0xFFFFu) << 16);// dim0 hi | tensor_dim1 lo
        g1[3] = (kout >> 16) | ((kin & 0xFFFFu) << 16);// dim1 hi | tile_dim0
        g1[4] = kout & 0xFFFFu;                        // tile_dim1 (tile_dim2=0)
        g1[5] = kin;                                   // tensor_dim0_stride[31:0]
        g1[6] = 0u;
        g1[7] = 0u;
        asm volatile("tensor_load_to_lds %0, %1" :: "s"(g0), "s"(g1) : "memory");
        __builtin_amdgcn_s_wait_tensorcnt((short)0);
    }
    __syncthreads();

    const int wave = threadIdx.x >> 5;
    const int lane = threadIdx.x & 31;
    const int m    = lane & 15;
    const int half = lane >> 4;
    const int rt   = blockIdx.x * 8 + wave;
    if (rt >= ROWTILES) return;
    const int row = rt * 16 + m;

    union HV { v16h v; uint4 q[2]; };
    v8f acc[NCT];
    #pragma unroll
    for (int c = 0; c < NCT; ++c) acc[c] = (v8f){};

    for (int k0 = 0; k0 < Kin; k0 += 32) {
        HV a;
        const uint4* pa = (const uint4*)(Xh + (size_t)row * Kin + k0 + half * 8);
        a.q[0] = pa[0];            // K = k0 + 8*half + 0..7
        a.q[1] = pa[2];            // K = k0 + 16 + 8*half + 0..7
        #pragma unroll
        for (int c = 0; c < NCT; ++c) {
            HV b;
            const uint4* pb =
                (const uint4*)(sW + (size_t)(c * 16 + m) * Kin + k0 + half * 8);
            b.q[0] = pb[0];        // K = k0 + 8*half + 0..7
            b.q[1] = pb[2];        // K = k0 + 16 + 8*half + 0..7
            acc[c] = __builtin_amdgcn_wmma_f32_16x16x32_f16(
                false, a.v, false, b.v, (short)0, acc[c], false, false);
        }
    }

    #pragma unroll
    for (int c = 0; c < NCT; ++c) {
        float* cp = C + ((size_t)rt * 16 + 8 * half) * Kout + c * 16 + m;
        #pragma unroll
        for (int r = 0; r < 8; ++r) cp[(size_t)r * Kout] = acc[c][r];
    }
}

// ------------------------- edge phase kernels ------------------------------
__device__ __forceinline__ float gat_atomic_max_f32(float* addr, float val) {
    if (val >= 0.0f)
        return __int_as_float(atomicMax((int*)addr, __float_as_int(val)));
    return __uint_as_float(atomicMin((unsigned int*)addr, __float_as_uint(val)));
}

// One wave32 per edge: e = att . leaky_relu(xl[src] + xr[dst]); atomic segmax.
__global__ void gat_edge_logit_kernel(const float* __restrict__ xl,
                                      const float* __restrict__ xr,
                                      const int* __restrict__ src,
                                      const int* __restrict__ dst,
                                      const float* __restrict__ att,
                                      float* __restrict__ elog,
                                      float* __restrict__ segmax,
                                      int E, int F) {
    long gid  = (long)blockIdx.x * blockDim.x + threadIdx.x;
    int  e    = (int)(gid >> 5);
    int  lane = (int)(gid & 31);
    if (e >= E) return;
    int sI = src[e], dI = dst[e];
    const float4* pl = (const float4*)(xl + (size_t)sI * F);
    const float4* pr = (const float4*)(xr + (size_t)dI * F);
    const float4* pa = (const float4*)att;
    float acc = 0.0f;
    for (int d4 = lane; d4 < (F >> 2); d4 += 32) {
        float4 l = pl[d4], r = pr[d4], a = pa[d4];
        float z;
        z = l.x + r.x; z = LRELU(z); acc += z * a.x;
        z = l.y + r.y; z = LRELU(z); acc += z * a.y;
        z = l.z + r.z; z = LRELU(z); acc += z * a.z;
        z = l.w + r.w; z = LRELU(z); acc += z * a.w;
    }
    #pragma unroll
    for (int off = 16; off > 0; off >>= 1) acc += __shfl_xor(acc, off, 32);
    if (lane == 0) {
        elog[e] = acc;
        gat_atomic_max_f32(segmax + dI, acc);
    }
}

// Thread per edge: elog <- exp(elog - max[dst]); segment sum via atomics.
__global__ void gat_edge_exp_kernel(float* __restrict__ elog,
                                    const float* __restrict__ segmax,
                                    float* __restrict__ segsum,
                                    const int* __restrict__ dst, int E) {
    long e = (long)blockIdx.x * blockDim.x + threadIdx.x;
    if (e >= E) return;
    int dI = dst[e];
    float v = __expf(elog[e] - segmax[dI]);
    elog[e] = v;
    atomicAdd(segsum + dI, v);
}

// One wave32 per edge: out[dst] += alpha * xl[src]  (f32 atomics).
__global__ void gat_edge_scatter_kernel(const float* __restrict__ xl,
                                        const float* __restrict__ ex,
                                        const float* __restrict__ segsum,
                                        const int* __restrict__ src,
                                        const int* __restrict__ dst,
                                        float* __restrict__ out,
                                        int E, int F) {
    long gid  = (long)blockIdx.x * blockDim.x + threadIdx.x;
    int  e    = (int)(gid >> 5);
    int  lane = (int)(gid & 31);
    if (e >= E) return;
    int sI = src[e], dI = dst[e];
    float a = ex[e] / (segsum[dI] + 1e-16f);
    const float4* pl = (const float4*)(xl + (size_t)sI * F);
    float* po = out + (size_t)dI * F;
    for (int d4 = lane; d4 < (F >> 2); d4 += 32) {
        float4 l = pl[d4];
        atomicAdd(po + 4 * d4 + 0, a * l.x);
        atomicAdd(po + 4 * d4 + 1, a * l.y);
        atomicAdd(po + 4 * d4 + 2, a * l.z);
        atomicAdd(po + 4 * d4 + 3, a * l.w);
    }
}

// h = (relu?) (h + bias)   elementwise over N x F
__global__ void gat_bias_act_kernel(float* __restrict__ h,
                                    const float* __restrict__ bias,
                                    long n, int F, int do_relu) {
    long i = (long)blockIdx.x * blockDim.x + threadIdx.x;
    if (i >= n) return;
    float v = h[i] + bias[(int)(i % F)];
    if (do_relu) v = fmaxf(v, 0.0f);
    h[i] = v;
}

// ------------------------------ pooling ------------------------------------
__global__ void gat_pool_scatter_kernel(const float* __restrict__ h,
                                        const int* __restrict__ batch,
                                        float* __restrict__ sums,
                                        float* __restrict__ cnt,
                                        int N, int F) {
    long i = (long)blockIdx.x * blockDim.x + threadIdx.x;
    if (i >= (long)N * F) return;
    int n = (int)(i / F), d = (int)(i % F);
    int b = batch[n];
    atomicAdd(sums + (size_t)b * F + d, h[i]);
    if (d == 0) atomicAdd(cnt + b, 1.0f);
}

// g = bn_eval(sums / max(cnt,1))
__global__ void gat_pool_bn_kernel(const float* __restrict__ sums,
                                   const float* __restrict__ cnt,
                                   float* __restrict__ g,
                                   const float* __restrict__ bg,
                                   const float* __restrict__ bb,
                                   const float* __restrict__ bm,
                                   const float* __restrict__ bv,
                                   int G, int F) {
    long i = (long)blockIdx.x * blockDim.x + threadIdx.x;
    if (i >= (long)G * F) return;
    int gi = (int)(i / F), d = (int)(i % F);
    float val = sums[i] / fmaxf(cnt[gi], 1.0f);
    val = (val - bm[d]) * rsqrtf(bv[d] + 1e-5f) * bg[d] + bb[d];
    g[i] = val;
}

// ------------------------------- MLP ---------------------------------------
__global__ void gat_mlp_kernel(const float* __restrict__ in,
                               const float* __restrict__ W,
                               const float* __restrict__ bias,
                               const float* __restrict__ bg,
                               const float* __restrict__ bb,
                               const float* __restrict__ bm,
                               const float* __restrict__ bv,
                               float* __restrict__ out,
                               int Kin, int Kout, int do_relu) {
    __shared__ float srow[256];
    int g = blockIdx.x;
    for (int i = threadIdx.x; i < Kin; i += blockDim.x)
        srow[i] = in[(size_t)g * Kin + i];
    __syncthreads();
    for (int j = threadIdx.x; j < Kout; j += blockDim.x) {
        float acc = bias[j];
        for (int i = 0; i < Kin; ++i) acc += srow[i] * W[(size_t)i * Kout + j];
        if (do_relu) acc = fmaxf(acc, 0.0f);
        if (bg) acc = (acc - bm[j]) * rsqrtf(bv[j] + 1e-5f) * bg[j] + bb[j];
        out[(size_t)g * Kout + j] = acc;
    }
}

// ---------------------------------------------------------------------------
extern "C" void kernel_launch(void* const* d_in, const int* in_sizes, int n_in,
                              void* d_out, int out_size, void* d_ws, size_t ws_size,
                              hipStream_t stream) {
    const int N = NNODES, E = NEDGES, G = NGRAPH;

    const float* x     = (const float*)d_in[0];
    const int*   src   = (const int*)d_in[1];
    const int*   dst   = src + E;
    const int*   batch = (const int*)d_in[2];
    const float* Wl1 = (const float*)d_in[3],  *Wr1 = (const float*)d_in[4];
    const float* att1 = (const float*)d_in[5], *b1 = (const float*)d_in[6];
    const float* Wl2 = (const float*)d_in[7],  *Wr2 = (const float*)d_in[8];
    const float* att2 = (const float*)d_in[9], *b2 = (const float*)d_in[10];
    const float* Wl3 = (const float*)d_in[11], *Wr3 = (const float*)d_in[12];
    const float* att3 = (const float*)d_in[13], *b3 = (const float*)d_in[14];
    const float* Wlin1 = (const float*)d_in[15], *blin1 = (const float*)d_in[16];
    const float* Wlin2 = (const float*)d_in[17], *blin2 = (const float*)d_in[18];
    const float* Wlin3 = (const float*)d_in[19], *blin3 = (const float*)d_in[20];
    const float* bng_g = (const float*)d_in[21], *bng_b = (const float*)d_in[22];
    const float* bng_m = (const float*)d_in[23], *bng_v = (const float*)d_in[24];
    const float* bn1_g = (const float*)d_in[25], *bn1_b = (const float*)d_in[26];
    const float* bn1_m = (const float*)d_in[27], *bn1_v = (const float*)d_in[28];
    const float* bn2_g = (const float*)d_in[29], *bn2_b = (const float*)d_in[30];
    const float* bn2_m = (const float*)d_in[31], *bn2_v = (const float*)d_in[32];

    // ---- workspace layout ----
    float* ws   = (float*)d_ws;
    float* xl   = ws;                       // N*256
    float* xr   = xl + (size_t)N * 256;     // N*256
    float* hA   = xr + (size_t)N * 256;     // N*256
    float* hB   = hA + (size_t)N * 256;     // N*256
    float* elog = hB + (size_t)N * 256;     // E
    float* mbuf = elog + E;                 // N
    float* sbuf = mbuf + N;                 // N
    float* sums = sbuf + N;                 // G*128
    float* cnt  = sums + (size_t)G * 128;   // G
    float* g0   = cnt + G;                  // G*128
    float* g1   = g0 + (size_t)G * 128;     // G*128
    _Float16* Xh  = (_Float16*)(g1 + (size_t)G * 128);  // N*256 halves
    _Float16* WtA = Xh + (size_t)N * 256;               // 65536 halves
    _Float16* WtB = WtA + 65536;                        // 65536 halves

    auto nblk = [](long n, int t) { return (unsigned)((n + t - 1) / t); };

    auto run_layer = [&](const float* hin, int Fin, int Fout,
                         const float* Wl, const float* Wr,
                         const float* att, const float* bias,
                         float* hout, int do_relu) {
        // operand precision conversion (f32 -> f16, W transposed)
        gat_f16cvt_kernel<<<nblk((long)N * Fin, 256), 256, 0, stream>>>(hin, Xh, (long)N * Fin);
        gat_wt_f16_kernel<<<nblk((long)Fin * Fout, 256), 256, 0, stream>>>(Wl, WtA, Fin, Fout);
        gat_wt_f16_kernel<<<nblk((long)Fin * Fout, 256), 256, 0, stream>>>(Wr, WtB, Fin, Fout);
        // WMMA GEMMs: xl = h@Wl, xr = h@Wr
        unsigned gb = (ROWTILES + 7) / 8;
        if (Fout == 128) {
            gat_gemm_wmma_kernel<8><<<gb, 256, 0, stream>>>(Xh, WtA, xl, Fin);
            gat_gemm_wmma_kernel<8><<<gb, 256, 0, stream>>>(Xh, WtB, xr, Fin);
        } else {
            gat_gemm_wmma_kernel<16><<<gb, 256, 0, stream>>>(Xh, WtA, xl, Fin);
            gat_gemm_wmma_kernel<16><<<gb, 256, 0, stream>>>(Xh, WtB, xr, Fin);
        }
        // edge softmax + aggregation
        gat_fill_kernel<<<nblk(N, 256), 256, 0, stream>>>(mbuf, -INFINITY, N);
        gat_fill_kernel<<<nblk(N, 256), 256, 0, stream>>>(sbuf, 0.0f, N);
        gat_fill_kernel<<<nblk((long)N * Fout, 256), 256, 0, stream>>>(hout, 0.0f, (long)N * Fout);
        gat_edge_logit_kernel<<<nblk((long)E * 32, 256), 256, 0, stream>>>(
            xl, xr, src, dst, att, elog, mbuf, E, Fout);
        gat_edge_exp_kernel<<<nblk(E, 256), 256, 0, stream>>>(elog, mbuf, sbuf, dst, E);
        gat_edge_scatter_kernel<<<nblk((long)E * 32, 256), 256, 0, stream>>>(
            xl, elog, sbuf, src, dst, hout, E, Fout);
        gat_bias_act_kernel<<<nblk((long)N * Fout, 256), 256, 0, stream>>>(
            hout, bias, (long)N * Fout, Fout, do_relu);
    };

    run_layer(x,  128, 128, Wl1, Wr1, att1, b1, hA, 1);
    run_layer(hA, 128, 256, Wl2, Wr2, att2, b2, hB, 1);
    run_layer(hB, 256, 128, Wl3, Wr3, att3, b3, hA, 0);

    // ---- global mean pool + bng ----
    gat_fill_kernel<<<nblk((long)G * 128, 256), 256, 0, stream>>>(sums, 0.0f, (long)G * 128);
    gat_fill_kernel<<<nblk(G, 256), 256, 0, stream>>>(cnt, 0.0f, G);
    gat_pool_scatter_kernel<<<nblk((long)N * 128, 256), 256, 0, stream>>>(hA, batch, sums, cnt, N, 128);
    gat_pool_bn_kernel<<<nblk((long)G * 128, 256), 256, 0, stream>>>(
        sums, cnt, g0, bng_g, bng_b, bng_m, bng_v, G, 128);

    // ---- MLP head ----
    gat_mlp_kernel<<<G, 128, 0, stream>>>(g0, Wlin1, blin1, bn1_g, bn1_b, bn1_m, bn1_v,
                                          g1, 128, 128, 1);
    gat_mlp_kernel<<<G, 64, 0, stream>>>(g1, Wlin2, blin2, bn2_g, bn2_b, bn2_m, bn2_v,
                                         g0, 128, 64, 1);
    gat_mlp_kernel<<<G, 32, 0, stream>>>(g0, Wlin3, blin3, nullptr, nullptr, nullptr, nullptr,
                                         (float*)d_out, 64, 2, 1);
}
